// HierarchicalTransformer_8203387535546
// MI455X (gfx1250) — matrix operand order repaired and checked by reading, not compiled, not run on hardware
//
#include <hip/hip_runtime.h>

#define NTOT   53248
#define NHOPS  4096
#define HOPLEN 12
#define DIMK   256
#define FFD    1024

typedef __attribute__((ext_vector_type(16))) __bf16 v16bf;
typedef __attribute__((ext_vector_type(8)))  __bf16 v8bf;
typedef __attribute__((ext_vector_type(8)))  float  v8f;

__device__ __forceinline__ __bf16 f2bf(float f) {
  unsigned u = __builtin_bit_cast(unsigned, f);
  u += 0x7fffu + ((u >> 16) & 1u);               // round-to-nearest-even
  unsigned short h = (unsigned short)(u >> 16);
  return __builtin_bit_cast(__bf16, h);
}

// ---------------------------------------------------------------- converters
__global__ void cvt_bf16(const float* __restrict__ src, __bf16* __restrict__ dst, int n) {
  int i = blockIdx.x * blockDim.x + threadIdx.x;
  if (i < n) dst[i] = f2bf(src[i]);
}

// dst[n*K + k] = src[k*Ncol + n]   (transpose to N-major bf16 for WMMA B)
__global__ void trc_bf16(const float* __restrict__ src, __bf16* __restrict__ dst,
                         int K, int Ncol) {
  int i = blockIdx.x * blockDim.x + threadIdx.x;
  if (i >= K * Ncol) return;
  int n = i / K, k = i % K;
  dst[(size_t)n * K + k] = f2bf(src[(size_t)k * Ncol + n]);
}

// Ws[k*8+h] = sum_c W[k,h*256+c]*att_src[h,c]; same for Wd with att_dst
__global__ void build_swd(const float* __restrict__ W, const float* __restrict__ ats,
                          const float* __restrict__ atd, float* __restrict__ Ws,
                          float* __restrict__ Wd) {
  int i = blockIdx.x * blockDim.x + threadIdx.x;
  if (i >= DIMK * 8) return;
  int k = i >> 3, h = i & 7;
  const float* wr = W + (size_t)k * 2048 + h * 256;
  const float* as = ats + h * 256;
  const float* ad = atd + h * 256;
  float s = 0.f, d = 0.f;
  for (int c = 0; c < 256; ++c) { float wv = wr[c]; s += wv * as[c]; d += wv * ad[c]; }
  Ws[i] = s; Wd[i] = d;
}

// Wagg[c*2048 + h*256 + k] = W[k*2048 + h*256 + c] / 8   (per-head transpose, N-major)
__global__ void build_wagg(const float* __restrict__ W, __bf16* __restrict__ Wagg) {
  int i = blockIdx.x * blockDim.x + threadIdx.x;
  if (i >= 256 * 2048) return;
  int c = i >> 11, r = i & 2047;
  int h = r >> 8, k = r & 255;
  Wagg[i] = f2bf(0.125f * W[(size_t)k * 2048 + h * 256 + c]);
}

// Wmt[c*256 + k] = (1/8) sum_h W[k*2048 + h*256 + c]
__global__ void build_wmt(const float* __restrict__ W, __bf16* __restrict__ Wmt) {
  int i = blockIdx.x * blockDim.x + threadIdx.x;
  if (i >= 256 * 256) return;
  int c = i >> 8, k = i & 255;
  float s = 0.f;
  for (int h = 0; h < 8; ++h) s += W[(size_t)k * 2048 + h * 256 + c];
  Wmt[i] = f2bf(0.125f * s);
}

// a_s[n,h] = x[n,:] @ Ws[:,h] ; a_d likewise
__global__ void attn_coef(const float* __restrict__ x, const float* __restrict__ Ws,
                          const float* __restrict__ Wd, float* __restrict__ a_s,
                          float* __restrict__ a_d) {
  int i = blockIdx.x * blockDim.x + threadIdx.x;
  if (i >= NTOT * 8) return;
  int n = i >> 3, h = i & 7;
  const float* xr = x + (size_t)n * DIMK;
  float s = 0.f, d = 0.f;
  for (int k = 0; k < DIMK; ++k) { float xv = xr[k]; s += xv * Ws[k * 8 + h]; d += xv * Wd[k * 8 + h]; }
  a_s[i] = s; a_d[i] = d;
}

// Per hop token: softmax over {self(x2), 12 members} per head; build weighted
// x-combination  xtb[j, h*256 + k] = sum_e alpha[e,h] * x[src_e, k]  (bf16)
__global__ void hop_attn(const float* __restrict__ x, const int* __restrict__ ei,
                         const float* __restrict__ a_s, const float* __restrict__ a_d,
                         __bf16* __restrict__ xtb) {
  int j = blockIdx.x;
  int tid = threadIdx.x;           // 256
  __shared__ float w[8][13];       // [head][0]=self weight (x2 multiplicity), 1..12 members
  __shared__ int   srcs[12];
  if (tid < HOPLEN) srcs[tid] = ei[NHOPS + j * HOPLEN + tid];
  __syncthreads();
  if (tid < 8) {
    int h = tid;
    float adj = a_d[j * 8 + h];
    float es = a_s[j * 8 + h] + adj; es = es > 0.f ? es : 0.2f * es;
    float em[HOPLEN]; float m = es;
    for (int t = 0; t < HOPLEN; ++t) {
      float e = a_s[srcs[t] * 8 + h] + adj; e = e > 0.f ? e : 0.2f * e;
      em[t] = e; m = fmaxf(m, e);
    }
    float wself = 2.f * __expf(es - m);    // self edge appears twice (explicit + added loop)
    float denom = wself;
    for (int t = 0; t < HOPLEN; ++t) { float e = __expf(em[t] - m); w[h][t + 1] = e; denom += e; }
    float inv = 1.f / (denom + 1e-16f);
    w[h][0] = wself * inv;
    for (int t = 0; t < HOPLEN; ++t) w[h][t + 1] *= inv;
  }
  __syncthreads();
  int k = tid;
  float xself = x[(size_t)j * DIMK + k];
  float xs[HOPLEN];
  for (int t = 0; t < HOPLEN; ++t) xs[t] = x[(size_t)srcs[t] * DIMK + k];
  for (int h = 0; h < 8; ++h) {
    float acc = w[h][0] * xself;
    #pragma unroll
    for (int t = 0; t < HOPLEN; ++t) acc += w[h][t + 1] * xs[t];
    xtb[(size_t)j * 2048 + h * 256 + k] = f2bf(acc);
  }
}

// ----------------------------------------------------------------- WMMA GEMM
// C[M,Nc] = A[M,K](bf16 row-major) x Bt[Nc,K](bf16 N-major), f32 accumulate.
// One wave computes a 16 x (TN*16) strip. blockDim = 256 (8 waves).
template <int TN>
__global__ void gemm_bf16(const __bf16* __restrict__ A, const __bf16* __restrict__ Bt,
                          const float* __restrict__ bias, float* __restrict__ outF,
                          __bf16* __restrict__ outB, int M, int K, int Nc, int relu) {
  const int lane = threadIdx.x & 31;
  const int wave = threadIdx.x >> 5;
  const int m0 = blockIdx.y * 16;
  const int n0 = (blockIdx.x * 8 + wave) * (TN * 16);
  if (m0 >= M || n0 >= Nc) return;          // wave-uniform: EXEC stays all-ones

  const int l15 = lane & 15;
  const int hi  = lane >> 4;

  v8f acc[TN] = {};
  const __bf16* arow = A + (size_t)(m0 + l15) * K + (hi ? 16 : 0);

  for (int kk = 0; kk < K; kk += 32) {
    v8bf a0 = *(const v8bf*)(arow + kk);
    v8bf a1 = *(const v8bf*)(arow + kk + 8);
    v16bf a = __builtin_shufflevector(a0, a1, 0, 1, 2, 3, 4, 5, 6, 7,
                                      8, 9, 10, 11, 12, 13, 14, 15);
    __builtin_prefetch(arow + kk + 64, 0, 1);
    #pragma unroll
    for (int t = 0; t < TN; ++t) {
      const __bf16* bp = Bt + (size_t)(n0 + t * 16 + l15) * K + kk + (hi ? 16 : 0);
      v16bf b = *(const v16bf*)bp;
      acc[t] = __builtin_amdgcn_wmma_f32_16x16x32_bf16(
          false, a, false, b, (short)0, acc[t], false, false);
    }
  }

  #pragma unroll
  for (int t = 0; t < TN; ++t) {
    int col = n0 + t * 16 + l15;
    float bv = bias ? bias[col] : 0.f;
    #pragma unroll
    for (int r = 0; r < 8; ++r) {
      int row = m0 + r + (hi ? 8 : 0);
      float v = acc[t][r] + bv;
      if (relu) v = fmaxf(v, 0.f);
      size_t o = (size_t)row * Nc + col;
      if (outF) outF[o] = v;
      if (outB) outB[o] = f2bf(v);
    }
  }
}

// --------------------------------------------- bias + residual + layer norm
// One wave per row of 256; lane holds 8 contiguous elements.
__global__ void bias_res_ln(const float* __restrict__ gin, const float* __restrict__ res,
                            const float* __restrict__ bias, const float* __restrict__ gam,
                            const float* __restrict__ bet, float* __restrict__ outF,
                            __bf16* __restrict__ outB, int rows) {
  int wv = (blockIdx.x * blockDim.x + threadIdx.x) >> 5;
  int lane = threadIdx.x & 31;
  if (wv >= rows) return;
  const float* gr = gin + (size_t)wv * DIMK;
  const float* rr = res + (size_t)wv * DIMK;
  float v[8]; float s = 0.f;
  #pragma unroll
  for (int i = 0; i < 8; ++i) {
    int c = lane * 8 + i;
    v[i] = gr[c] + rr[c] + (bias ? bias[c] : 0.f);
    s += v[i];
  }
  #pragma unroll
  for (int off = 16; off; off >>= 1) s += __shfl_xor(s, off, 32);
  float mu = s * (1.f / 256.f);
  float q = 0.f;
  #pragma unroll
  for (int i = 0; i < 8; ++i) { float d = v[i] - mu; q += d * d; }
  #pragma unroll
  for (int off = 16; off; off >>= 1) q += __shfl_xor(q, off, 32);
  float rstd = rsqrtf(q * (1.f / 256.f) + 1e-5f);
  #pragma unroll
  for (int i = 0; i < 8; ++i) {
    int c = lane * 8 + i;
    float o = (v[i] - mu) * rstd * gam[c] + bet[c];
    if (outF) outF[(size_t)wv * DIMK + c] = o;
    if (outB) outB[(size_t)wv * DIMK + c] = f2bf(o);
  }
}

// ---------------------------------------------------------------------------
extern "C" void kernel_launch(void* const* d_in, const int* in_sizes, int n_in,
                              void* d_out, int out_size, void* d_ws, size_t ws_size,
                              hipStream_t stream) {
  const float* x    = (const float*)d_in[0];
  const int*   ei   = (const int*)  d_in[1];
  const float* gatW = (const float*)d_in[2];
  const float* atS  = (const float*)d_in[3];
  const float* atD  = (const float*)d_in[4];
  const float* gatB = (const float*)d_in[5];
  const float* W1   = (const float*)d_in[6];
  const float* b1   = (const float*)d_in[7];
  const float* W2   = (const float*)d_in[8];
  const float* b2   = (const float*)d_in[9];
  const float* ln1g = (const float*)d_in[10];
  const float* ln1b = (const float*)d_in[11];
  const float* ln2g = (const float*)d_in[12];
  const float* ln2b = (const float*)d_in[13];
  float* out = (float*)d_out;

  char* base = (char*)d_ws;
  size_t off = 0;
  auto alloc = [&](size_t bytes) { void* p = base + off; off = (off + bytes + 255) & ~(size_t)255; return p; };

  __bf16* xb   = (__bf16*)alloc((size_t)NTOT * DIMK * 2);
  float*  Ws   = (float*) alloc(DIMK * 8 * 4);
  float*  Wd   = (float*) alloc(DIMK * 8 * 4);
  float*  a_s  = (float*) alloc((size_t)NTOT * 8 * 4);
  float*  a_d  = (float*) alloc((size_t)NTOT * 8 * 4);
  __bf16* Wagg = (__bf16*)alloc((size_t)256 * 2048 * 2);
  __bf16* Wmt  = (__bf16*)alloc((size_t)256 * 256 * 2);
  __bf16* Wt1  = (__bf16*)alloc((size_t)FFD * DIMK * 2);
  __bf16* Wt2  = (__bf16*)alloc((size_t)DIMK * FFD * 2);
  float*  g0   = (float*) alloc((size_t)NTOT * DIMK * 4);   // reused as y2
  __bf16* xtb  = (__bf16*)alloc((size_t)NHOPS * 2048 * 2);
  float*  x1   = (float*) alloc((size_t)NTOT * DIMK * 4);
  __bf16* x1b  = (__bf16*)alloc((size_t)NTOT * DIMK * 2);
  __bf16* y1b  = (__bf16*)alloc((size_t)NTOT * FFD * 2);
  float*  y2   = g0;
  (void)ws_size; (void)in_sizes; (void)n_in; (void)out_size;

  // Weight prep + activation conversion
  { int n = NTOT * DIMK; cvt_bf16<<<(n + 255) / 256, 256, 0, stream>>>(x, xb, n); }
  build_swd<<<(DIMK * 8 + 255) / 256, 256, 0, stream>>>(gatW, atS, atD, Ws, Wd);
  build_wagg<<<(256 * 2048) / 256, 256, 0, stream>>>(gatW, Wagg);
  build_wmt<<<(256 * 256) / 256, 256, 0, stream>>>(gatW, Wmt);
  trc_bf16<<<(DIMK * FFD) / 256, 256, 0, stream>>>(W1, Wt1, DIMK, FFD);
  trc_bf16<<<(FFD * DIMK) / 256, 256, 0, stream>>>(W2, Wt2, FFD, DIMK);

  // Attention coefficients a_src/a_dst for all nodes
  attn_coef<<<(NTOT * 8 + 255) / 256, 256, 0, stream>>>(x, Ws, Wd, a_s, a_d);

  // Member-node (degree-1) GAT output for all rows: g0 = x @ mean_h(W_h)
  gemm_bf16<2><<<dim3(1, NTOT / 16), 256, 0, stream>>>(xb, Wmt, nullptr, g0, nullptr,
                                                       NTOT, DIMK, DIMK, 0);
  // Hop tokens: attention-weighted feature combos, then per-head GEMM (fused as K=2048)
  hop_attn<<<NHOPS, 256, 0, stream>>>(x, ei, a_s, a_d, xtb);
  gemm_bf16<2><<<dim3(1, NHOPS / 16), 256, 0, stream>>>(xtb, Wagg, nullptr, g0, nullptr,
                                                        NHOPS, 2048, DIMK, 0);
  // x1 = LN1(g0 + gat_bias + x)
  bias_res_ln<<<(NTOT + 7) / 8, 256, 0, stream>>>(g0, x, gatB, ln1g, ln1b, x1, x1b, NTOT);

  // FFN
  gemm_bf16<4><<<dim3(FFD / 512, NTOT / 16), 256, 0, stream>>>(x1b, Wt1, b1, nullptr, y1b,
                                                               NTOT, DIMK, FFD, 1);
  gemm_bf16<2><<<dim3(1, NTOT / 16), 256, 0, stream>>>(y1b, Wt2, b2, y2, nullptr,
                                                       NTOT, FFD, DIMK, 0);
  // out = LN2(y2 + x1)
  bias_res_ln<<<(NTOT + 7) / 8, 256, 0, stream>>>(y2, x1, nullptr, ln2g, ln2b, out, nullptr, NTOT);
}
